// NerDetector_89567247991530
// MI455X (gfx1250) — matrix operand order repaired and checked by reading, not compiled, not run on hardware
//
#include <hip/hip_runtime.h>
#include <hip/hip_bf16.h>
#include <math.h>

// Problem constants (reference: B,T,W,D,NT = 32,512,250,768,2)
#define BB 32
#define TT 512
#define WW 250
#define DD 768
#define G4 3072   // 4*D gate rows
#define H2 1536   // 2*D bi-directional hidden

typedef __attribute__((ext_vector_type(16))) __bf16 v16bf;
typedef __attribute__((ext_vector_type(8)))  __bf16 v8bf;
typedef __attribute__((ext_vector_type(8)))  float  v8f;

union ABu { v16bf v; v8bf h[2]; };

__device__ __forceinline__ v8bf ld8(const __bf16* p) {
  return *reinterpret_cast<const v8bf*>(p);
}

// ---------------------------------------------------------------------------
// 1) first-subtoken gather: feat[b,w,:] = bert_out[b, first_idx(b,w), :] (bf16)
// ---------------------------------------------------------------------------
__global__ __launch_bounds__(256) void k_gather(const float* __restrict__ bert,
                                                const int* __restrict__ wids,
                                                __bf16* __restrict__ feat) {
  int b = blockIdx.x / WW, w = blockIdx.x % WW;
  __shared__ int smin;
  if (threadIdx.x == 0) smin = TT - 1;
  __syncthreads();
  for (int t = threadIdx.x; t < TT; t += 256)
    if (wids[b * TT + t] == w) atomicMin(&smin, t);
  __syncthreads();
  int idx = smin;
  const float* src = bert + ((size_t)b * TT + idx) * DD;
  __bf16* dst = feat + ((size_t)b * WW + w) * DD;
  for (int d = threadIdx.x; d < DD; d += 256) dst[d] = (__bf16)src[d];
}

// ---------------------------------------------------------------------------
// 2) build bf16 concat weights: wcat[g, 0:768] = w_hh[g,:], wcat[g, 768:] = w_ih[g,:]
// ---------------------------------------------------------------------------
__global__ void k_build_wcat(const float* __restrict__ whh,
                             const float* __restrict__ wih,
                             int Din, __bf16* __restrict__ out) {
  const int K = DD + Din;
  const size_t total = (size_t)G4 * K;
  for (size_t i = blockIdx.x * (size_t)blockDim.x + threadIdx.x; i < total;
       i += (size_t)gridDim.x * blockDim.x) {
    int row = (int)(i / K), k = (int)(i % K);
    float v = (k < DD) ? whh[(size_t)row * DD + k]
                       : wih[(size_t)row * Din + (k - DD)];
    out[i] = (__bf16)v;
  }
}

// ---------------------------------------------------------------------------
// 3) fused LSTM step: g = [h_prev ; x_t] @ wcat^T + b, then cell update.
//    One wave = one (mtile, ntile, dir) 16x16 hidden tile, 4 gate accumulators.
//    48 blocks x 128 threads = 192 waves (2 dirs x 2 mtiles x 48 ntiles).
// ---------------------------------------------------------------------------
__global__ __launch_bounds__(128) void k_lstm_step(
    const __bf16* __restrict__ x, int Din,
    const __bf16* __restrict__ wf, const __bf16* __restrict__ wr,
    const float* __restrict__ bf_, const float* __restrict__ br_,
    __bf16* __restrict__ hbuf,   // (2 dir, 2 parity, 32, 768) bf16
    float* __restrict__ cbuf,    // (2 dir, 32, 768) f32
    __bf16* __restrict__ seq_bf, // (32, 250, 1536) bf16 or null
    float* __restrict__ seq_f32, // (32, 250, 1536) f32 or null
    int step) {
  const int K = DD + Din;
  const int lane = threadIdx.x & 31;
  const int wv = threadIdx.x >> 5;
  const int gw = blockIdx.x * 4 + wv;
  const int dir = gw / 96;
  const int rem = gw % 96;
  const int mtile = rem / 48;
  const int ntile = rem % 48;
  const int half = lane >> 4;
  const int t = dir ? (WW - 1 - step) : step;

  const __bf16* w = dir ? wr : wf;
  const float* bias = dir ? br_ : bf_;
  const __bf16* hprev = hbuf + (size_t)(dir * 2 + (step & 1)) * BB * DD;
  __bf16* hnext = hbuf + (size_t)(dir * 2 + ((step + 1) & 1)) * BB * DD;
  float* c = cbuf + (size_t)dir * BB * DD;

  const int am = mtile * 16 + (lane & 15);  // A-matrix row (batch), per ISA A layout
  const int bn = ntile * 16 + (lane & 15);  // B-matrix column (hidden index)

  v8f acc[4] = {};  // i, f, g, o accumulators (f32)

  for (int k0 = 0; k0 < K; k0 += 32) {
    // A tile 16x32 bf16: K slices [half*8, half*8+8) and [16+half*8, ...)
    const __bf16* asrc = (k0 < DD)
        ? (hprev + (size_t)am * DD + k0)
        : (x + ((size_t)am * WW + t) * (size_t)Din + (k0 - DD));
    ABu a;
    a.h[0] = ld8(asrc + half * 8);
    a.h[1] = ld8(asrc + 16 + half * 8);
#pragma unroll
    for (int q = 0; q < 4; q++) {
      // B tile 32x16 bf16: K slices [half*8, half*8+8) and [16+half*8, ...) at row (q*768 + bn)
      const __bf16* bsrc = w + (size_t)(q * DD + bn) * K + k0 + half * 8;
      ABu bm;
      bm.h[0] = ld8(bsrc);
      bm.h[1] = ld8(bsrc + 16);
      acc[q] = __builtin_amdgcn_wmma_f32_16x16x32_bf16(
          false, a.v, false, bm.v, (short)0, acc[q], false, false);
    }
    __builtin_prefetch(asrc + 32, 0, 1);  // global_prefetch_b8 for next K tile
  }

  // cell update; C/D layout: VGPR r -> row r + 8*half, col = lane&15
  const int n = ntile * 16 + (lane & 15);
#pragma unroll
  for (int r = 0; r < 8; r++) {
    int batch = mtile * 16 + r + half * 8;
    float gi = acc[0][r] + bias[0 * DD + n];
    float gf = acc[1][r] + bias[1 * DD + n];
    float gg = acc[2][r] + bias[2 * DD + n];
    float go = acc[3][r] + bias[3 * DD + n];
    size_t hi = (size_t)batch * DD + n;
    float co = c[hi];
    float si = 1.f / (1.f + __expf(-gi));
    float sf = 1.f / (1.f + __expf(-gf));
    float so = 1.f / (1.f + __expf(-go));
    float cn = sf * co + si * tanhf(gg);
    float h = so * tanhf(cn);
    c[hi] = cn;
    hnext[hi] = (__bf16)h;
    size_t oi = ((size_t)batch * WW + t) * H2 + (size_t)dir * DD + n;
    if (seq_bf)  seq_bf[oi] = (__bf16)h;
    if (seq_f32) seq_f32[oi] = h;
  }
}

// ---------------------------------------------------------------------------
// 4) linear (NT=2) + softmax: one wave per (b,w) row
// ---------------------------------------------------------------------------
__global__ __launch_bounds__(256) void k_linear_softmax(
    const float* __restrict__ h2, const float* __restrict__ wlin,
    const float* __restrict__ blin, float* __restrict__ prob) {
  int row = blockIdx.x * 8 + (threadIdx.x >> 5);
  int lane = threadIdx.x & 31;
  if (row >= BB * WW) return;
  const float* hr = h2 + (size_t)row * H2;
  float d0 = 0.f, d1 = 0.f;
  for (int k = lane; k < H2; k += 32) {
    float hv = hr[k];
    d0 += hv * wlin[k];
    d1 += hv * wlin[H2 + k];
  }
  for (int m = 16; m; m >>= 1) {
    d0 += __shfl_xor(d0, m, 32);
    d1 += __shfl_xor(d1, m, 32);
  }
  if (lane == 0) {
    float z0 = d0 + blin[0], z1 = d1 + blin[1];
    float mx = fmaxf(z0, z1);
    float e0 = __expf(z0 - mx), e1 = __expf(z1 - mx);
    float s = e0 + e1;
    prob[(size_t)row * 2 + 0] = e0 / s;
    prob[(size_t)row * 2 + 1] = e1 / s;
  }
}

// ---------------------------------------------------------------------------
// 5) CRF log-likelihood (NT=2): one lane per batch, wave-reduce mean
// ---------------------------------------------------------------------------
__device__ __forceinline__ float lse2(float x, float y) {
  float m = fmaxf(x, y);
  return m + __logf(__expf(x - m) + __expf(y - m));
}

__global__ void k_crf_llh(const float* __restrict__ prob,
                          const int* __restrict__ tags,
                          const float* __restrict__ st,
                          const float* __restrict__ en,
                          const float* __restrict__ tr,
                          float* __restrict__ loss) {
  int b = threadIdx.x;  // 32 lanes, one wave
  const float* e = prob + (size_t)b * WW * 2;
  const int* tg = tags + (size_t)b * WW;
  float num = st[tg[0]] + en[tg[WW - 1]];
  for (int t = 0; t < WW; t++) num += e[t * 2 + tg[t]];
  for (int t = 0; t < WW - 1; t++) num += tr[tg[t] * 2 + tg[t + 1]];
  float a0 = st[0] + e[0], a1 = st[1] + e[1];
  for (int t = 1; t < WW; t++) {
    float n0 = lse2(a0 + tr[0], a1 + tr[2]) + e[t * 2 + 0];
    float n1 = lse2(a0 + tr[1], a1 + tr[3]) + e[t * 2 + 1];
    a0 = n0; a1 = n1;
  }
  float v = num - lse2(a0 + en[0], a1 + en[1]);
  for (int m = 16; m; m >>= 1) v += __shfl_xor(v, m, 32);
  if (b == 0) loss[0] = -(v / (float)BB);
}

// ---------------------------------------------------------------------------
// 6) Viterbi decode (NT=2): one lane per batch
// ---------------------------------------------------------------------------
__global__ void k_viterbi(const float* __restrict__ prob,
                          const float* __restrict__ st,
                          const float* __restrict__ en,
                          const float* __restrict__ tr,
                          signed char* __restrict__ bp,
                          float* __restrict__ path) {
  int b = threadIdx.x;
  const float* e = prob + (size_t)b * WW * 2;
  signed char* bpb = bp + (size_t)b * (WW - 1) * 2;
  float s0 = st[0] + e[0], s1 = st[1] + e[1];
  for (int t = 1; t < WW; t++) {
    float c00 = s0 + tr[0], c10 = s1 + tr[2];
    int i0 = (c10 > c00);                 // argmax, first index wins ties
    float m0 = fmaxf(c00, c10);
    float c01 = s0 + tr[1], c11 = s1 + tr[3];
    int i1 = (c11 > c01);
    float m1 = fmaxf(c01, c11);
    bpb[(t - 1) * 2 + 0] = (signed char)i0;
    bpb[(t - 1) * 2 + 1] = (signed char)i1;
    s0 = m0 + e[t * 2 + 0];
    s1 = m1 + e[t * 2 + 1];
  }
  int cur = ((s1 + en[1]) > (s0 + en[0])) ? 1 : 0;
  path[(size_t)b * WW + WW - 1] = (float)cur;
  for (int t = WW - 2; t >= 0; t--) {
    cur = bpb[t * 2 + cur];
    path[(size_t)b * WW + t] = (float)cur;
  }
}

// ---------------------------------------------------------------------------
extern "C" void kernel_launch(void* const* d_in, const int* in_sizes, int n_in,
                              void* d_out, int out_size, void* d_ws, size_t ws_size,
                              hipStream_t stream) {
  const float* bert      = (const float*)d_in[0];
  const int*   wids      = (const int*)d_in[1];
  const int*   tags      = (const int*)d_in[2];
  const float* w_ih_l0   = (const float*)d_in[3];
  const float* w_hh_l0   = (const float*)d_in[4];
  const float* b_l0      = (const float*)d_in[5];
  const float* w_ih_l0r  = (const float*)d_in[6];
  const float* w_hh_l0r  = (const float*)d_in[7];
  const float* b_l0r     = (const float*)d_in[8];
  const float* w_ih_l1   = (const float*)d_in[9];
  const float* w_hh_l1   = (const float*)d_in[10];
  const float* b_l1      = (const float*)d_in[11];
  const float* w_ih_l1r  = (const float*)d_in[12];
  const float* w_hh_l1r  = (const float*)d_in[13];
  const float* b_l1r     = (const float*)d_in[14];
  const float* w_lin     = (const float*)d_in[15];
  const float* b_lin     = (const float*)d_in[16];
  const float* crf_start = (const float*)d_in[17];
  const float* crf_end   = (const float*)d_in[18];
  const float* crf_trans = (const float*)d_in[19];

  float* out      = (float*)d_out;
  float* path_out = out;                      // [0, 8000)
  float* prob_out = out + BB * WW;            // [8000, 24000)
  float* loss_out = out + BB * WW + BB * WW * 2;  // [24000]

  size_t off = 0;
  auto alloc = [&](size_t bytes) -> char* {
    char* p = (char*)d_ws + off;
    off += (bytes + 255) & ~(size_t)255;
    return p;
  };
  __bf16* feat   = (__bf16*)alloc((size_t)BB * WW * DD * 2);
  __bf16* wc_l0f = (__bf16*)alloc((size_t)G4 * (DD + DD) * 2);
  __bf16* wc_l0r = (__bf16*)alloc((size_t)G4 * (DD + DD) * 2);
  __bf16* wc_l1f = (__bf16*)alloc((size_t)G4 * (DD + H2) * 2);
  __bf16* wc_l1r = (__bf16*)alloc((size_t)G4 * (DD + H2) * 2);
  __bf16* h1     = (__bf16*)alloc((size_t)BB * WW * H2 * 2);
  float*  h2     = (float*) alloc((size_t)BB * WW * H2 * 4);
  __bf16* hbuf   = (__bf16*)alloc((size_t)4 * BB * DD * 2);
  float*  cbuf   = (float*) alloc((size_t)2 * BB * DD * 4);
  signed char* bp = (signed char*)alloc((size_t)BB * (WW - 1) * 2);

  // bf16 weight concat [w_hh | w_ih] per layer/direction
  k_build_wcat<<<1024, 256, 0, stream>>>(w_hh_l0,  w_ih_l0,  DD, wc_l0f);
  k_build_wcat<<<1024, 256, 0, stream>>>(w_hh_l0r, w_ih_l0r, DD, wc_l0r);
  k_build_wcat<<<1024, 256, 0, stream>>>(w_hh_l1,  w_ih_l1,  H2, wc_l1f);
  k_build_wcat<<<1024, 256, 0, stream>>>(w_hh_l1r, w_ih_l1r, H2, wc_l1r);
  k_gather<<<BB * WW, 256, 0, stream>>>(bert, wids, feat);

  // layer 0 (bi-directional, fused per-step WMMA GEMM)
  hipMemsetAsync(hbuf, 0, (size_t)4 * BB * DD * 2, stream);
  hipMemsetAsync(cbuf, 0, (size_t)2 * BB * DD * 4, stream);
  for (int s = 0; s < WW; s++)
    k_lstm_step<<<48, 128, 0, stream>>>(feat, DD, wc_l0f, wc_l0r, b_l0, b_l0r,
                                        hbuf, cbuf, h1, nullptr, s);

  // layer 1
  hipMemsetAsync(hbuf, 0, (size_t)4 * BB * DD * 2, stream);
  hipMemsetAsync(cbuf, 0, (size_t)2 * BB * DD * 4, stream);
  for (int s = 0; s < WW; s++)
    k_lstm_step<<<48, 128, 0, stream>>>(h1, H2, wc_l1f, wc_l1r, b_l1, b_l1r,
                                        hbuf, cbuf, nullptr, h2, s);

  k_linear_softmax<<<(BB * WW) / 8, 256, 0, stream>>>(h2, w_lin, b_lin, prob_out);
  k_crf_llh<<<1, 32, 0, stream>>>(prob_out, tags, crf_start, crf_end, crf_trans, loss_out);
  k_viterbi<<<1, 32, 0, stream>>>(prob_out, crf_start, crf_end, crf_trans, bp, path_out);
}
